// GCN_30219389894763
// MI455X (gfx1250) — matrix operand order repaired and checked by reading, not compiled, not run on hardware
//
#include <hip/hip_runtime.h>

#define F_IN 512
#define HID  16
#define NCLS 7

typedef __attribute__((ext_vector_type(16))) __bf16 v16bf;
typedef __attribute__((ext_vector_type(8)))  float  v8f;

union BF16V {
    v16bf v;
    unsigned int w[8];
};

__device__ __forceinline__ void atomic_addf(float* p, float v) {
    __hip_atomic_fetch_add(p, v, __ATOMIC_RELAXED, __HIP_MEMORY_SCOPE_AGENT);
}

// ---- degree: self-loop contributes 1 ----
__global__ void k_deg_init(float* __restrict__ deg, int n) {
    int i = blockIdx.x * blockDim.x + threadIdx.x;
    if (i < n) deg[i] = 1.0f;
}

__global__ void k_deg_edges(const int* __restrict__ col, float* __restrict__ deg, int e) {
    int i = blockIdx.x * blockDim.x + threadIdx.x;
    if (i < e) atomic_addf(&deg[col[i]], 1.0f);
}

__global__ void k_dinv(float* __restrict__ deg, int n) {
    int i = blockIdx.x * blockDim.x + threadIdx.x;
    if (i < n) deg[i] = rsqrtf(deg[i]);   // deg >= 1 always (self loop)
}

__global__ void k_zero(float* __restrict__ p, long long n) {
    long long i = (long long)blockIdx.x * blockDim.x + threadIdx.x;
    if (i < n) p[i] = 0.0f;
}

// ---- GEMM1: t1[n,16] = x[n,512] @ W1[512,16] via v_wmma_f32_16x16x32_bf16 ----
// Block = 256 threads = 8 waves; each wave owns a 16-row x 16-col tile.
__global__ __launch_bounds__(256) void k_gemm1(const float* __restrict__ x,
                                               const float* __restrict__ W1,
                                               float* __restrict__ t1, int n) {
    // W1 staged to LDS as bf16, n-major with padded stride (bank-conflict relief).
    __shared__ __bf16 w1t[HID * 520];
    int tid = threadIdx.x;
    for (int idx = tid; idx < F_IN * HID; idx += 256) {
        int k  = idx >> 4;      // W1 is [k][nn] row-major
        int nn = idx & 15;
        w1t[nn * 520 + k] = (__bf16)W1[idx];   // native v_cvt bf16
    }
    __syncthreads();

    int wave = tid >> 5;
    int lane = tid & 31;
    int m    = lane & 15;      // row-in-tile (A) / column n (B,C)
    int kh   = lane >> 4;      // half-select in K per ISA layout
    int tileBase = (blockIdx.x * 8 + wave) * 16;
    if (tileBase >= n) return;                   // wave-uniform: whole tile past end

    int rowA = tileBase + m;
    if (rowA >= n) rowA = n - 1;                 // clamp loads (tail tiles only)
    const float* xrow = x + (size_t)rowA * F_IN;
    const __bf16* brow = &w1t[m * 520];

    v8f acc = {};
    for (int k0 = 0; k0 < F_IN; k0 += 32) {
        __builtin_prefetch(xrow + k0 + 128, 0, 1);   // global_prefetch_b8, next chunk
        // A operand: lane needs 16 consecutive K values [k0+kh*16, +16)
        const float4* pa0 = (const float4*)(xrow + k0 + kh * 16);
        const float4* pa1 = (const float4*)(xrow + k0 + kh * 16 + 8);
        float4 q0 = pa0[0], q1 = pa0[1], q2 = pa1[0], q3 = pa1[1];
        v16bf a;
        a[0]  = (__bf16)q0.x; a[1]  = (__bf16)q0.y;
        a[2]  = (__bf16)q0.z; a[3]  = (__bf16)q0.w;
        a[4]  = (__bf16)q1.x; a[5]  = (__bf16)q1.y;
        a[6]  = (__bf16)q1.z; a[7]  = (__bf16)q1.w;
        a[8]  = (__bf16)q2.x; a[9]  = (__bf16)q2.y;
        a[10] = (__bf16)q2.z; a[11] = (__bf16)q2.w;
        a[12] = (__bf16)q3.x; a[13] = (__bf16)q3.y;
        a[14] = (__bf16)q3.z; a[15] = (__bf16)q3.w;

        // B operand: 16 consecutive K values (bf16) at [k0 + kh*16]
        BF16V b;
        const unsigned int* pb = (const unsigned int*)(brow + k0 + kh * 16);
#pragma unroll
        for (int j = 0; j < 8; ++j) b.w[j] = pb[j];

        acc = __builtin_amdgcn_wmma_f32_16x16x32_bf16(
            false, a, false, b.v, (short)0, acc, false, false);
    }

    // C layout: VGPR j -> row tileBase + j + kh*8, col m
    if (tileBase + 15 < n) {
        float* trow = t1 + (size_t)(tileBase + kh * 8) * HID + m;
#pragma unroll
        for (int j = 0; j < 8; ++j) trow[(size_t)j * HID] = acc[j];
    } else {
#pragma unroll
        for (int j = 0; j < 8; ++j) {
            int r = tileBase + j + kh * 8;
            if (r < n) t1[(size_t)r * HID + m] = acc[j];
        }
    }
}

// ---- layer-1 edge aggregation: one thread per (edge, feature) ----
__global__ void k_agg1(const int* __restrict__ row, const int* __restrict__ col,
                       const float* __restrict__ dinv, const float* __restrict__ t1,
                       float* __restrict__ agg, long long total) {
    long long t = (long long)blockIdx.x * blockDim.x + threadIdx.x;
    if (t >= total) return;
    int e = (int)(t >> 4);
    int f = (int)(t & 15);
    int r = row[e], c = col[e];
    float w = dinv[r] * dinv[c];
    atomic_addf(&agg[(size_t)c * HID + f], w * t1[(size_t)r * HID + f]);
}

// ---- self-loop + bias + relu (agg updated in place -> h) ----
__global__ void k_relu(float* __restrict__ agg, const float* __restrict__ t1,
                       const float* __restrict__ dinv, const float* __restrict__ b1,
                       long long total) {
    long long t = (long long)blockIdx.x * blockDim.x + threadIdx.x;
    if (t >= total) return;
    int i = (int)(t >> 4);
    int f = (int)(t & 15);
    float di = dinv[i];
    float v = agg[t] + di * di * t1[t] + b1[f];
    agg[t] = v > 0.0f ? v : 0.0f;
}

// ---- GEMM2: t2[n,7] = h[n,16] @ W2[16,7] (tiny K: scalar FMA) ----
__global__ void k_gemm2(const float* __restrict__ h, const float* __restrict__ W2,
                        float* __restrict__ t2, int n) {
    int i = blockIdx.x * blockDim.x + threadIdx.x;
    if (i >= n) return;
    const float4* hp = (const float4*)(h + (size_t)i * HID);
    float4 h0 = hp[0], h1 = hp[1], h2 = hp[2], h3 = hp[3];
    float hv[16] = {h0.x, h0.y, h0.z, h0.w, h1.x, h1.y, h1.z, h1.w,
                    h2.x, h2.y, h2.z, h2.w, h3.x, h3.y, h3.z, h3.w};
#pragma unroll
    for (int c = 0; c < NCLS; ++c) {
        float s = 0.0f;
#pragma unroll
        for (int f = 0; f < HID; ++f) s = fmaf(hv[f], W2[f * NCLS + c], s);
        t2[(size_t)i * NCLS + c] = s;
    }
}

// ---- output init: bias + self-loop term ----
__global__ void k_out_init(const float* __restrict__ t2, const float* __restrict__ dinv,
                           const float* __restrict__ b2, float* __restrict__ out, int n) {
    long long t = (long long)blockIdx.x * blockDim.x + threadIdx.x;
    int i = (int)(t >> 3);
    int c = (int)(t & 7);
    if (i >= n || c >= NCLS) return;
    float di = dinv[i];
    out[(size_t)i * NCLS + c] = b2[c] + di * di * t2[(size_t)i * NCLS + c];
}

// ---- layer-2 edge aggregation ----
__global__ void k_agg2(const int* __restrict__ row, const int* __restrict__ col,
                       const float* __restrict__ dinv, const float* __restrict__ t2,
                       float* __restrict__ out, long long total) {
    long long t = (long long)blockIdx.x * blockDim.x + threadIdx.x;
    if (t >= total) return;
    int e = (int)(t >> 3);
    int c = (int)(t & 7);
    if (c >= NCLS) return;
    int r = row[e], cc = col[e];
    float w = dinv[r] * dinv[cc];
    atomic_addf(&out[(size_t)cc * NCLS + c], w * t2[(size_t)r * NCLS + c]);
}

extern "C" void kernel_launch(void* const* d_in, const int* in_sizes, int n_in,
                              void* d_out, int out_size, void* d_ws, size_t ws_size,
                              hipStream_t stream) {
    const float* x  = (const float*)d_in[0];
    const int*   ei = (const int*)d_in[1];
    const float* W1 = (const float*)d_in[2];
    const float* b1 = (const float*)d_in[3];
    const float* W2 = (const float*)d_in[4];
    const float* b2 = (const float*)d_in[5];
    float* out = (float*)d_out;

    int n = in_sizes[0] / F_IN;     // 100000
    int e = in_sizes[1] / 2;        // 3200000
    const int* row = ei;            // edge_index[0] (source)
    const int* col = ei + e;        // edge_index[1] (target)

    // workspace layout
    char* ws = (char*)d_ws;
    float* deg = (float*)ws;                                  // n floats (becomes dinv)
    size_t off = ((size_t)n * sizeof(float) + 255) & ~(size_t)255;
    float* t1  = (float*)(ws + off);                          // n*16
    float* agg = t1 + (size_t)n * HID;                        // n*16 (becomes h)
    float* t2  = t1;                                          // reuse: t1 dead after k_relu

    const int B = 256;
    long long nh   = (long long)n * HID;
    long long tot1 = (long long)e * HID;
    long long tot2 = (long long)e * 8;
    int ntiles = (n + 15) / 16;

    k_deg_init <<<(n + B - 1) / B, B, 0, stream>>>(deg, n);
    k_deg_edges<<<(e + B - 1) / B, B, 0, stream>>>(col, deg, e);
    k_dinv     <<<(n + B - 1) / B, B, 0, stream>>>(deg, n);

    k_gemm1    <<<(ntiles + 7) / 8, 256, 0, stream>>>(x, W1, t1, n);
    k_zero     <<<(unsigned)((nh + B - 1) / B), B, 0, stream>>>(agg, nh);
    k_agg1     <<<(unsigned)((tot1 + B - 1) / B), B, 0, stream>>>(row, col, deg, t1, agg, tot1);
    k_relu     <<<(unsigned)((nh + B - 1) / B), B, 0, stream>>>(agg, t1, deg, b1, nh);

    k_gemm2    <<<(n + B - 1) / B, B, 0, stream>>>(agg, W2, t2, n);
    k_out_init <<<(unsigned)(((long long)n * 8 + B - 1) / B), B, 0, stream>>>(t2, deg, b2, out, n);
    k_agg2     <<<(unsigned)((tot2 + B - 1) / B), B, 0, stream>>>(row, col, deg, t2, out, tot2);
}
